// StockGAT_2911987826951
// MI455X (gfx1250) — matrix operand order repaired and checked by reading, not compiled, not run on hardware
//
#include <hip/hip_runtime.h>
#include <hip/hip_bf16.h>

typedef __attribute__((ext_vector_type(16))) _Float16 v16h;
typedef __attribute__((ext_vector_type(8)))  float    v8f;

#define NEG_SLOPE 0.2f

// ---------------------------------------------------------------------------
// f32 -> f16 with zero padding:  dst[rowsPad, colsPad] = pad(src[rows, cols])
// ---------------------------------------------------------------------------
__global__ void cvt_pad_f16_kernel(const float* __restrict__ src, _Float16* __restrict__ dst,
                                   int rows, int cols, int rowsPad, int colsPad) {
    size_t idx = (size_t)blockIdx.x * blockDim.x + threadIdx.x;
    size_t tot = (size_t)rowsPad * colsPad;
    if (idx >= tot) return;
    int r = (int)(idx / colsPad);
    int c = (int)(idx % colsPad);
    float v = (r < rows && c < cols) ? src[(size_t)r * cols + c] : 0.0f;
    dst[idx] = (_Float16)v;
}

// ---------------------------------------------------------------------------
// Pack weight f16 [Kpad, M] (row-major) into the per-lane WMMA B layout.
// 16-bit B (32x16 per k-step): lanes 0-15 hold K = k0..k0+7, k0+16..k0+23,
// lanes 16-31 hold K = k0+8..k0+15, k0+24..k0+31, column = lane & 15.
// Bp[((ct*KT + kt)*32 + lane)*16 + i] = W[(kbase + (i&7) + ((i>>3)<<4))*M + col]
// ---------------------------------------------------------------------------
__global__ void pack_b_kernel(const _Float16* __restrict__ W16, _Float16* __restrict__ Bp,
                              int Kpad, int M) {
    int idx = blockIdx.x * blockDim.x + threadIdx.x;
    int KT = Kpad >> 5;
    int CT = M >> 4;
    int total = CT * KT * 32;
    if (idx >= total) return;
    int lane = idx & 31;
    int kt = (idx >> 5) % KT;
    int ct = (idx >> 5) / KT;
    int col = (ct << 4) + (lane & 15);
    int kbase = (kt << 5) + ((lane >> 4) << 3);
    _Float16* dst = Bp + (size_t)idx * 16;
#pragma unroll
    for (int i = 0; i < 16; ++i)
        dst[i] = W16[(size_t)(kbase + (i & 7) + ((i >> 3) << 4)) * M + col];
}

// ---------------------------------------------------------------------------
// WMMA GEMM: C[Nrows, M] f32 = A[Npad, Kpad] f16  x  Bpacked, M == 512.
// Block = 256 threads = 8 waves; block tile = 32 rows x 512 cols.
// Wave w owns cols [w*64, w*64+64), rows [0,32) -> 8 WMMA accumulators.
// A k-step tile (32 rows x 32 halves = 2 KB) is staged into LDS with
// async global->LDS DMA (double-buffered, waves 0..3 issue one b128 DMA per
// lane per k-step), so the 8 waves share one copy of A and the next k-step's
// A fetch overlaps the current WMMA burst.
// ---------------------------------------------------------------------------
__global__ void __launch_bounds__(256)
wmma_gemm_kernel(const _Float16* __restrict__ A, const _Float16* __restrict__ Bp,
                 float* __restrict__ C, int Kpad, int M, int Nrows) {
    __shared__ alignas(16) _Float16 sA[2][32 * 32];

    const int wave = threadIdx.x >> 5;
    const int lane = threadIdx.x & 31;
    const int row0 = blockIdx.x << 5;
    const int col0 = wave << 6;          // 8 waves * 64 cols = 512
    const int KT = Kpad >> 5;
    const int r = lane & 15;
    const int kb = (lane >> 4) << 3;     // 0 or 8 halves
    const int rowhi = (lane >> 4) << 3;  // C/D row offset 0 or 8

    v8f acc[8];
#pragma unroll
    for (int i = 0; i < 8; ++i) acc[i] = (v8f){};

    // waves 0..3 each DMA 8 rows (lane covers 16B) of the 32x32-half A tile
    auto stage_issue = [&](int buf, int k0) {
        if (wave < 4) {
            int rr = (wave << 3) + (lane >> 2);   // tile row 0..31
            int cc = (lane & 3) << 3;             // half offset 0,8,16,24
            unsigned lds = (unsigned)(uintptr_t)&sA[buf][rr * 32 + cc];
            const _Float16* g = A + (size_t)(row0 + rr) * Kpad + k0 + cc;
            asm volatile("global_load_async_to_lds_b128 %0, %1, off"
                         :: "v"(lds), "v"((unsigned long long)(uintptr_t)g)
                         : "memory");
        }
    };
    auto stage_wait = [&]() {
        if (wave < 4) asm volatile("s_wait_asynccnt 0" ::: "memory");
        __syncthreads();
    };

    stage_issue(0, 0);
    stage_wait();

    for (int kt = 0; kt < KT; ++kt) {
        const int buf = kt & 1;
        const _Float16* sa = &sA[buf][0];

        // A fragments from LDS (16-bit A 16x32 layout; tile row stride = 32 halves)
        union { v16h v; float4 f[2]; } a0, a1;
        a0.f[0] = *(const float4*)(sa + r * 32 + kb);
        a0.f[1] = *(const float4*)(sa + r * 32 + 16 + kb);
        a1.f[0] = *(const float4*)(sa + (16 + r) * 32 + kb);
        a1.f[1] = *(const float4*)(sa + (16 + r) * 32 + 16 + kb);

        // all B fragments for this k-step up front (one wait amortized over 8 WMMAs)
        union { v16h v; float4 f[2]; } b0, b1, b2, b3;
        {
            const size_t base = (size_t)(col0 >> 4) * KT + kt;
            const _Float16* l0 = Bp + ((base + 0 * KT) * 32 + lane) * 16;
            const _Float16* l1 = Bp + ((base + 1 * KT) * 32 + lane) * 16;
            const _Float16* l2 = Bp + ((base + 2 * KT) * 32 + lane) * 16;
            const _Float16* l3 = Bp + ((base + 3 * KT) * 32 + lane) * 16;
            b0.f[0] = *(const float4*)(l0); b0.f[1] = *(const float4*)(l0 + 8);
            b1.f[0] = *(const float4*)(l1); b1.f[1] = *(const float4*)(l1 + 8);
            b2.f[0] = *(const float4*)(l2); b2.f[1] = *(const float4*)(l2 + 8);
            b3.f[0] = *(const float4*)(l3); b3.f[1] = *(const float4*)(l3 + 8);
        }

        if (kt + 1 < KT) stage_issue(buf ^ 1, (kt + 1) << 5);

        acc[0] = __builtin_amdgcn_wmma_f32_16x16x32_f16(false, a0.v, false, b0.v, (short)0, acc[0], false, false);
        acc[4] = __builtin_amdgcn_wmma_f32_16x16x32_f16(false, a1.v, false, b0.v, (short)0, acc[4], false, false);
        acc[1] = __builtin_amdgcn_wmma_f32_16x16x32_f16(false, a0.v, false, b1.v, (short)0, acc[1], false, false);
        acc[5] = __builtin_amdgcn_wmma_f32_16x16x32_f16(false, a1.v, false, b1.v, (short)0, acc[5], false, false);
        acc[2] = __builtin_amdgcn_wmma_f32_16x16x32_f16(false, a0.v, false, b2.v, (short)0, acc[2], false, false);
        acc[6] = __builtin_amdgcn_wmma_f32_16x16x32_f16(false, a1.v, false, b2.v, (short)0, acc[6], false, false);
        acc[3] = __builtin_amdgcn_wmma_f32_16x16x32_f16(false, a0.v, false, b3.v, (short)0, acc[3], false, false);
        acc[7] = __builtin_amdgcn_wmma_f32_16x16x32_f16(false, a1.v, false, b3.v, (short)0, acc[7], false, false);

        if (kt + 1 < KT) stage_wait();
    }

    // C/D layout: VGPR rr, lane<16 -> (row rr, col lane); lane>=16 -> row rr+8
#pragma unroll
    for (int half = 0; half < 2; ++half) {
#pragma unroll
        for (int j = 0; j < 4; ++j) {
            const v8f a = acc[half * 4 + j];
            const int cb = col0 + (j << 4) + (lane & 15);
#pragma unroll
            for (int rr = 0; rr < 8; ++rr) {
                int grow = row0 + half * 16 + rowhi + rr;
                if (grow < Nrows) C[(size_t)grow * M + cb] = a[rr];
            }
        }
    }
}

// ---------------------------------------------------------------------------
// alpha_s/alpha_d [N,4]: one wave per (node, head), reduce 128 channels.
// ---------------------------------------------------------------------------
__global__ void alpha_kernel(const float* __restrict__ h, const float* __restrict__ a_src,
                             const float* __restrict__ a_dst, float* __restrict__ as,
                             float* __restrict__ ad, int Nn) {
    int wid = (int)(((size_t)blockIdx.x * blockDim.x + threadIdx.x) >> 5);
    int lane = threadIdx.x & 31;
    if (wid >= Nn * 4) return;
    int n = wid >> 2, hh = wid & 3;
    const float* hp = h + (size_t)n * 512 + hh * 128;
    const float* asp = a_src + hh * 128;
    const float* adp = a_dst + hh * 128;
    float ss = 0.0f, sd = 0.0f;
    for (int c = lane; c < 128; c += 32) {
        float v = hp[c];
        ss += v * asp[c];
        sd += v * adp[c];
    }
    for (int o = 16; o; o >>= 1) {
        ss += __shfl_xor(ss, o, 32);
        sd += __shfl_xor(sd, o, 32);
    }
    if (lane == 0) { as[wid] = ss; ad[wid] = sd; }
}

// order-preserving float<->uint for atomic max over possibly-negative values
__device__ __forceinline__ unsigned fkey(float f) {
    unsigned u = __float_as_uint(f);
    return (u & 0x80000000u) ? ~u : (u | 0x80000000u);
}
__device__ __forceinline__ float fkeyinv(unsigned u) {
    unsigned b = (u & 0x80000000u) ? (u & 0x7fffffffu) : ~u;
    return __uint_as_float(b);
}

__device__ __forceinline__ void edge_sd(const int* __restrict__ ei, int E, int e,
                                        int& s, int& d) {
    if (e < E) { s = ei[e]; d = ei[E + e]; }
    else       { s = e - E; d = e - E; }   // implicit self loops
}

// pass 1: segment max of leaky_relu(alpha_s[src] + alpha_d[dst]) over dst
__global__ void edge_max_kernel(const int* __restrict__ ei, const float* __restrict__ as,
                                const float* __restrict__ ad, unsigned* __restrict__ mbits,
                                int E, int Et) {
    int idx = (int)((size_t)blockIdx.x * blockDim.x + threadIdx.x);
    if (idx >= Et * 4) return;
    int e = idx >> 2, hh = idx & 3;
    int s, d; edge_sd(ei, E, e, s, d);
    float v = as[s * 4 + hh] + ad[d * 4 + hh];
    v = v > 0.0f ? v : NEG_SLOPE * v;
    atomicMax(&mbits[d * 4 + hh], fkey(v));
}

// pass 2: numerators + segment sum of exp(e - m[dst])
__global__ void edge_exp_kernel(const int* __restrict__ ei, const float* __restrict__ as,
                                const float* __restrict__ ad, const unsigned* __restrict__ mbits,
                                float* __restrict__ denom, float* __restrict__ eexp,
                                int E, int Et) {
    int idx = (int)((size_t)blockIdx.x * blockDim.x + threadIdx.x);
    if (idx >= Et * 4) return;
    int e = idx >> 2, hh = idx & 3;
    int s, d; edge_sd(ei, E, e, s, d);
    float v = as[s * 4 + hh] + ad[d * 4 + hh];
    v = v > 0.0f ? v : NEG_SLOPE * v;
    float m = fkeyinv(mbits[d * 4 + hh]);
    float pv = __expf(v - m);
    eexp[idx] = pv;
    atomicAdd(&denom[d * 4 + hh], pv);
}

// pass 3: weighted scatter of h[src] into out[dst]; one wave per (edge, head),
// lane handles 4 contiguous channels (float4 load + 4 f32 atomics)
__global__ void edge_agg_kernel(const int* __restrict__ ei, const float* __restrict__ h,
                                const float* __restrict__ eexp, const float* __restrict__ denom,
                                float* __restrict__ out, int E, int Et) {
    int wid = (int)(((size_t)blockIdx.x * blockDim.x + threadIdx.x) >> 5);
    int lane = threadIdx.x & 31;
    if (wid >= Et * 4) return;
    int e = wid >> 2, hh = wid & 3;
    int s, d; edge_sd(ei, E, e, s, d);
    float w = eexp[wid] / (denom[d * 4 + hh] + 1e-16f);
    const float4 v = *(const float4*)(h + (size_t)s * 512 + hh * 128 + lane * 4);
    float* op = out + (size_t)d * 512 + hh * 128 + lane * 4;
    atomicAdd(op + 0, v.x * w);
    atomicAdd(op + 1, v.y * w);
    atomicAdd(op + 2, v.z * w);
    atomicAdd(op + 3, v.w * w);
}

// bias + relu in place on f32 aggregate, plus f16 copy (zero-padded rows) for next GEMM
__global__ void bias_relu_cvt_kernel(float* __restrict__ agg, const float* __restrict__ bias,
                                     _Float16* __restrict__ a16, int Nn, int Npad) {
    size_t idx = (size_t)blockIdx.x * blockDim.x + threadIdx.x;
    size_t tot = (size_t)Npad * 512;
    if (idx >= tot) return;
    int r = (int)(idx >> 9);
    int c = (int)(idx & 511);
    float v = 0.0f;
    if (r < Nn) {
        v = agg[idx] + bias[c];
        v = v > 0.0f ? v : 0.0f;
        agg[idx] = v;
    }
    a16[idx] = (_Float16)v;
}

// conv3: h3[N,4] = act[N,512] @ W3[512,4]  (tiny-N GEMM, W3 staged in LDS)
__global__ void gemm3_kernel(const float* __restrict__ act, const float* __restrict__ W3,
                             float* __restrict__ h3, int Nn) {
    __shared__ float w[512 * 4];
    for (int i = threadIdx.x; i < 2048; i += blockDim.x) w[i] = W3[i];
    __syncthreads();
    int idx = (int)((size_t)blockIdx.x * blockDim.x + threadIdx.x);
    if (idx >= Nn * 4) return;
    int n = idx >> 2, col = idx & 3;
    const float* a = act + (size_t)n * 512;
    float sum = 0.0f;
#pragma unroll 8
    for (int k = 0; k < 512; ++k) sum += a[k] * w[k * 4 + col];
    h3[idx] = sum;
}

// layer-3 attention coefficients (C == 1): alpha = h3 * a
__global__ void alpha3_kernel(const float* __restrict__ h3, const float* __restrict__ a3s,
                              const float* __restrict__ a3d, float* __restrict__ as,
                              float* __restrict__ ad, int Nn) {
    int idx = (int)((size_t)blockIdx.x * blockDim.x + threadIdx.x);
    if (idx >= Nn * 4) return;
    int hh = idx & 3;
    float v = h3[idx];
    as[idx] = v * a3s[hh];
    ad[idx] = v * a3d[hh];
}

// layer-3 aggregation (C == 1): out3[dst,h] += h3[src,h] * alpha
__global__ void edge_agg3_kernel(const int* __restrict__ ei, const float* __restrict__ h3,
                                 const float* __restrict__ eexp, const float* __restrict__ denom,
                                 float* __restrict__ out3, int E, int Et) {
    int idx = (int)((size_t)blockIdx.x * blockDim.x + threadIdx.x);
    if (idx >= Et * 4) return;
    int e = idx >> 2, hh = idx & 3;
    int s, d; edge_sd(ei, E, e, s, d);
    float w = eexp[idx] / (denom[d * 4 + hh] + 1e-16f);
    atomicAdd(&out3[d * 4 + hh], h3[s * 4 + hh] * w);
}

// mean over heads + bias -> final output [N,1]
__global__ void final_kernel(const float* __restrict__ out3, const float* __restrict__ b3,
                             float* __restrict__ out, int Nn) {
    int n = (int)((size_t)blockIdx.x * blockDim.x + threadIdx.x);
    if (n >= Nn) return;
    float s = out3[n * 4 + 0] + out3[n * 4 + 1] + out3[n * 4 + 2] + out3[n * 4 + 3];
    out[n] = 0.25f * s + b3[0];
}

// ---------------------------------------------------------------------------
extern "C" void kernel_launch(void* const* d_in, const int* in_sizes, int n_in,
                              void* d_out, int out_size, void* d_ws, size_t ws_size,
                              hipStream_t stream) {
    const float* x   = (const float*)d_in[0];
    const int*   ei  = (const int*)  d_in[1];
    const float* W1  = (const float*)d_in[2];
    const float* a1s = (const float*)d_in[3];
    const float* a1d = (const float*)d_in[4];
    const float* b1  = (const float*)d_in[5];
    const float* W2  = (const float*)d_in[6];
    const float* a2s = (const float*)d_in[7];
    const float* a2d = (const float*)d_in[8];
    const float* b2  = (const float*)d_in[9];
    const float* W3  = (const float*)d_in[10];
    const float* a3s = (const float*)d_in[11];
    const float* a3d = (const float*)d_in[12];
    const float* b3  = (const float*)d_in[13];
    float* out = (float*)d_out;

    const int N    = in_sizes[0] / 30;
    const int E    = in_sizes[1] / 2;
    const int Et   = E + N;                // implicit self loops
    const int Npad = (N + 31) & ~31;       // GEMM block tile = 32 rows
    const int M    = 512;
    const int T    = 256;
    const int ET4  = Et * 4;

    // workspace carve-up
    char* p = (char*)d_ws;
    auto carve = [&](size_t bytes) -> void* {
        void* r = (void*)p;
        p += (bytes + 255) & ~(size_t)255;
        return r;
    };
    _Float16* A16   = (_Float16*)carve((size_t)Npad * M * 2);
    float*    Hbuf  = (float*)   carve((size_t)Npad * M * 4);
    float*    Agg   = (float*)   carve((size_t)Npad * M * 4);
    _Float16* W16   = (_Float16*)carve((size_t)M * M * 2);
    _Float16* Bpk   = (_Float16*)carve((size_t)M * M * 2);
    float*    as_   = (float*)   carve((size_t)N * 16);
    float*    ad_   = (float*)   carve((size_t)N * 16);
    unsigned* mbits = (unsigned*)carve((size_t)N * 16);
    float*    denom = (float*)   carve((size_t)N * 16);
    float*    eexp  = (float*)   carve((size_t)Et * 16);
    float*    h3    = (float*)   carve((size_t)N * 16);
    float*    out3  = (float*)   carve((size_t)N * 16);
    (void)ws_size; (void)n_in; (void)out_size;

    auto edge_softmax = [&](const float* asv, const float* adv) {
        hipMemsetAsync(mbits, 0, (size_t)N * 16, stream);
        edge_max_kernel<<<(ET4 + T - 1) / T, T, 0, stream>>>(ei, asv, adv, mbits, E, Et);
        hipMemsetAsync(denom, 0, (size_t)N * 16, stream);
        edge_exp_kernel<<<(ET4 + T - 1) / T, T, 0, stream>>>(ei, asv, adv, mbits, denom, eexp, E, Et);
    };

    // ---------------- layer 1: 30 -> 4x128, concat ----------------
    {
        size_t atot = (size_t)Npad * 32;
        cvt_pad_f16_kernel<<<(unsigned)((atot + T - 1) / T), T, 0, stream>>>(x, A16, N, 30, Npad, 32);
        size_t wtot = (size_t)32 * M;
        cvt_pad_f16_kernel<<<(unsigned)((wtot + T - 1) / T), T, 0, stream>>>(W1, W16, 30, M, 32, M);
        int ptot = (M / 16) * (32 / 32) * 32;
        pack_b_kernel<<<(ptot + T - 1) / T, T, 0, stream>>>(W16, Bpk, 32, M);
        wmma_gemm_kernel<<<Npad / 32, T, 0, stream>>>(A16, Bpk, Hbuf, 32, M, N);

        alpha_kernel<<<(N * 4 + 7) / 8, T, 0, stream>>>(Hbuf, a1s, a1d, as_, ad_, N);
        edge_softmax(as_, ad_);
        hipMemsetAsync(Agg, 0, (size_t)N * M * 4, stream);
        edge_agg_kernel<<<(ET4 + 7) / 8, T, 0, stream>>>(ei, Hbuf, eexp, denom, Agg, E, Et);

        size_t btot = (size_t)Npad * M;
        bias_relu_cvt_kernel<<<(unsigned)((btot + T - 1) / T), T, 0, stream>>>(Agg, b1, A16, N, Npad);
    }

    // ---------------- layer 2: 512 -> 4x128, concat ----------------
    {
        size_t wtot = (size_t)M * M;
        cvt_pad_f16_kernel<<<(unsigned)((wtot + T - 1) / T), T, 0, stream>>>(W2, W16, M, M, M, M);
        int ptot = (M / 16) * (M / 32) * 32;
        pack_b_kernel<<<(ptot + T - 1) / T, T, 0, stream>>>(W16, Bpk, M, M);
        wmma_gemm_kernel<<<Npad / 32, T, 0, stream>>>(A16, Bpk, Hbuf, M, M, N);

        alpha_kernel<<<(N * 4 + 7) / 8, T, 0, stream>>>(Hbuf, a2s, a2d, as_, ad_, N);
        edge_softmax(as_, ad_);
        hipMemsetAsync(Agg, 0, (size_t)N * M * 4, stream);
        edge_agg_kernel<<<(ET4 + 7) / 8, T, 0, stream>>>(ei, Hbuf, eexp, denom, Agg, E, Et);

        size_t btot = (size_t)Npad * M;
        bias_relu_cvt_kernel<<<(unsigned)((btot + T - 1) / T), T, 0, stream>>>(Agg, b2, A16, N, Npad);
    }

    // ---------------- layer 3: 512 -> 4x1, mean over heads ----------------
    {
        gemm3_kernel<<<(N * 4 + T - 1) / T, T, 0, stream>>>(Agg, W3, h3, N);
        alpha3_kernel<<<(N * 4 + T - 1) / T, T, 0, stream>>>(h3, a3s, a3d, as_, ad_, N);
        edge_softmax(as_, ad_);
        hipMemsetAsync(out3, 0, (size_t)N * 16, stream);
        edge_agg3_kernel<<<(ET4 + T - 1) / T, T, 0, stream>>>(ei, h3, eexp, denom, out3, E, Et);
        final_kernel<<<(N + T - 1) / T, T, 0, stream>>>(out3, b3, out, N);
    }
}
